// MultiSelfAttentionHead_Varlen_Flex_50319836839973
// MI455X (gfx1250) — compile-verified, matrix operand order checked by reading, not run on hardware
//
#include <hip/hip_runtime.h>
#include <hip/hip_bf16.h>

typedef __attribute__((ext_vector_type(16))) _Float16 v16h;
typedef __attribute__((ext_vector_type(8)))  _Float16 v8h;
typedef __attribute__((ext_vector_type(4)))  _Float16 v4h;
typedef __attribute__((ext_vector_type(8)))  float    v8f;
typedef __attribute__((ext_vector_type(4)))  int      v4i;

#define DMODEL 1024
#define NTOK   2048
#define NHEAD  16
#define DHEAD  64
#define NACT   256   // active query rows (q_idx < 256; others fully masked -> 0)

// ---------------------------------------------------------------------------
// Fragment loader for 16-bit A/B operands of v_wmma_f32_16x16x32_f16.
// Per ISA layout (CDNA5 ISA §7.12.2): lanes 0-15 hold row/col L with
// K = {0..7, 16..23}; lanes 16-31 hold the same row/col with K = {8..15, 24..31}.
// Both chunks are contiguous 16-byte loads.
// ---------------------------------------------------------------------------
__device__ __forceinline__ v16h ldfrag(const _Float16* rowbase, int ld, int lane) {
  const int r  = lane & 15;
  const int kb = (lane & 16) ? 8 : 0;
  const _Float16* p = rowbase + (size_t)r * ld + kb;
  union { v16h v; v8h h[2]; } u;
  u.h[0] = *(const v8h*)(p);        // K = kb .. kb+7
  u.h[1] = *(const v8h*)(p + 16);   // K = kb+16 .. kb+23
  return u.v;
}

__device__ __forceinline__ v8f wmma_f16(v16h a, v16h b, v8f c) {
  return __builtin_amdgcn_wmma_f32_16x16x32_f16(false, a, false, b, (short)0, c,
                                                false, false);
}

// ---------------------------------------------------------------------------
// CDNA5 async global->LDS copy (16 B per lane, tracked by ASYNCcnt).
// Probe result: builtin takes (v4i __device__*, ..., imm, imm) — cast to the
// exact vector pointee + address spaces. Fallback is a plain VGPR round-trip.
// ---------------------------------------------------------------------------
#if __has_builtin(__builtin_amdgcn_global_load_async_to_lds_b128)
#define HAVE_ASYNC_LDS 1
typedef __attribute__((address_space(1))) v4i* v4i_gptr;
typedef __attribute__((address_space(3))) v4i* v4i_lptr;
#else
#define HAVE_ASYNC_LDS 0
#endif

__device__ __forceinline__ void async_g2l_b128(const void* g, void* l) {
#if HAVE_ASYNC_LDS
  __builtin_amdgcn_global_load_async_to_lds_b128((v4i_gptr)(g), (v4i_lptr)(l),
                                                 0, 0);
#else
  *(v8h*)l = *(const v8h*)g;
#endif
}

__device__ __forceinline__ void wait_async0() {
#if HAVE_ASYNC_LDS
#if __has_builtin(__builtin_amdgcn_s_wait_asynccnt)
  __builtin_amdgcn_s_wait_asynccnt(0);
#else
  asm volatile("s_wait_asynccnt 0" ::: "memory");
#endif
#endif
}

// ---------------------------------------------------------------------------
// K0: LayerNorm(x) over D=1024, write f16. One block (256 thr) per row.
// ---------------------------------------------------------------------------
__global__ __launch_bounds__(256) void ln_rows_kernel(const float* __restrict__ x,
                                                      _Float16* __restrict__ xn) {
  const int row = blockIdx.x;
  const int tid = threadIdx.x;
  const float4 v = ((const float4*)(x + (size_t)row * DMODEL))[tid];
  float s  = v.x + v.y + v.z + v.w;
  float ss = v.x * v.x + v.y * v.y + v.z * v.z + v.w * v.w;
#pragma unroll
  for (int d = 1; d < 32; d <<= 1) {
    s  += __shfl_xor(s,  d, 32);
    ss += __shfl_xor(ss, d, 32);
  }
  __shared__ float redS[8], redQ[8];
  if ((tid & 31) == 0) { redS[tid >> 5] = s; redQ[tid >> 5] = ss; }
  __syncthreads();
  float S = 0.f, Q = 0.f;
#pragma unroll
  for (int i = 0; i < 8; ++i) { S += redS[i]; Q += redQ[i]; }
  const float mu  = S * (1.f / DMODEL);
  const float var = Q * (1.f / DMODEL) - mu * mu;
  const float inv = rsqrtf(var + 1e-5f);
  v4h o;
  o[0] = (_Float16)((v.x - mu) * inv);
  o[1] = (_Float16)((v.y - mu) * inv);
  o[2] = (_Float16)((v.z - mu) * inv);
  o[3] = (_Float16)((v.w - mu) * inv);
  *(v4h*)(xn + (size_t)row * DMODEL + tid * 4) = o;
}

// ---------------------------------------------------------------------------
// K0w: convert+transpose weight f32[k][n] -> f16 Wt[n][k] (B fragments become
// contiguous row loads). 32x32 tiles, block (32,8).
// ---------------------------------------------------------------------------
__global__ __launch_bounds__(256) void wtrans_kernel(const float* __restrict__ w,
                                                     _Float16* __restrict__ wt) {
  __shared__ float tile[32][33];
  const int bx = blockIdx.x * 32;   // n
  const int by = blockIdx.y * 32;   // k
  const int tx = threadIdx.x, ty = threadIdx.y;
#pragma unroll
  for (int i = 0; i < 32; i += 8)
    tile[ty + i][tx] = w[(size_t)(by + ty + i) * DMODEL + (bx + tx)];
  __syncthreads();
#pragma unroll
  for (int i = 0; i < 32; i += 8)
    wt[(size_t)(bx + ty + i) * DMODEL + (by + tx)] = (_Float16)tile[tx][ty + i];
}

// ---------------------------------------------------------------------------
// K1: q/k/v projections. C[m][n] = sum_k xn[m][k] * Wt[n][k], f16 in, f32 acc,
// f16 out. Workgroup 128x128 tile, 4 waves of 64x64, grid.z selects weight.
// ---------------------------------------------------------------------------
__global__ __launch_bounds__(128) void gemm_qkv_kernel(const _Float16* __restrict__ xn,
                                                       const _Float16* __restrict__ wt,
                                                       _Float16* __restrict__ qkv) {
  const int lane = threadIdx.x & 31;
  const int wave = threadIdx.x >> 5;
  const int m0 = blockIdx.y * 128 + (wave >> 1) * 64;
  const int n0 = blockIdx.x * 128 + (wave & 1) * 64;
  const _Float16* A = xn;
  const _Float16* B = wt + (size_t)blockIdx.z * DMODEL * DMODEL;
  _Float16* C = qkv + (size_t)blockIdx.z * NTOK * DMODEL;

  v8f acc[4][4] = {};
  for (int kk = 0; kk < DMODEL; kk += 32) {
    if (kk + 32 < DMODEL) {
      __builtin_prefetch(A + (size_t)m0 * DMODEL + kk + 32, 0, 1);
      __builtin_prefetch(B + (size_t)n0 * DMODEL + kk + 32, 0, 1);
    }
    v16h a[4];
#pragma unroll
    for (int i = 0; i < 4; ++i)
      a[i] = ldfrag(A + (size_t)(m0 + 16 * i) * DMODEL + kk, DMODEL, lane);
#pragma unroll
    for (int j = 0; j < 4; ++j) {
      v16h b = ldfrag(B + (size_t)(n0 + 16 * j) * DMODEL + kk, DMODEL, lane);
#pragma unroll
      for (int i = 0; i < 4; ++i)
        acc[i][j] = wmma_f16(a[i], b, acc[i][j]);
    }
  }
  // C/D layout: VGPR j -> row j (lanes 0-15) / j+8 (lanes 16-31), col = lane&15
  const int col  = lane & 15;
  const int rofs = (lane & 16) ? 8 : 0;
#pragma unroll
  for (int i = 0; i < 4; ++i)
#pragma unroll
    for (int j = 0; j < 4; ++j)
#pragma unroll
      for (int jj = 0; jj < 8; ++jj)
        C[(size_t)(m0 + 16 * i + jj + rofs) * DMODEL + (n0 + 16 * j + col)] =
            (_Float16)acc[i][j][jj];
}

// ---------------------------------------------------------------------------
// K2: per-head LayerNorm (dh=64) in-place on q and k (f16). One wave per
// (row, head) segment; lane handles 2 elements.
// ---------------------------------------------------------------------------
__global__ __launch_bounds__(256) void head_ln_kernel(_Float16* __restrict__ q,
                                                      _Float16* __restrict__ k) {
  _Float16* t = (blockIdx.y == 0) ? q : k;
  const int lane = threadIdx.x & 31;
  const int wave = threadIdx.x >> 5;
  const int seg  = blockIdx.x * 8 + wave;       // 0 .. NTOK*NHEAD-1
  const int row  = seg >> 4;
  const int h    = seg & 15;
  _Float16* p = t + (size_t)row * DMODEL + h * DHEAD;
  float a = (float)p[lane];
  float b = (float)p[lane + 32];
  float s = a + b, ss = a * a + b * b;
#pragma unroll
  for (int d = 1; d < 32; d <<= 1) {
    s  += __shfl_xor(s,  d, 32);
    ss += __shfl_xor(ss, d, 32);
  }
  const float mu  = s * (1.f / DHEAD);
  const float var = ss * (1.f / DHEAD) - mu * mu;
  const float inv = rsqrtf(var + 1e-5f);
  p[lane]      = (_Float16)((a - mu) * inv);
  p[lane + 32] = (_Float16)((b - mu) * inv);
}

// ---------------------------------------------------------------------------
// K3: block-sparse attention. One wave per (head h, q-block r). The 16 q rows
// 16r..16r+15 attend exactly to kv rows {r + 16j : j=0..127}.
//   S[16,128] = Q Kt (16 WMMAs), dense softmax over the gathered 128 columns,
//   O[16,64] = P V (16 WMMAs via LDS re-fragmenting). V is staged to LDS with
//   CDNA5 async-to-LDS copies that overlap the S-phase WMMAs.
// ---------------------------------------------------------------------------
__global__ __launch_bounds__(32) void attention_kernel(const _Float16* __restrict__ q,
                                                       const _Float16* __restrict__ k,
                                                       const _Float16* __restrict__ v,
                                                       _Float16* __restrict__ aout) {
  const int h    = blockIdx.x & 15;
  const int r    = blockIdx.x >> 4;      // 0..15
  const int lane = threadIdx.x;

  __shared__ _Float16 p_lds[16][128];
  __shared__ _Float16 v_lds[128][DHEAD];

  // Kick off async staging of the V tile (gathered rows r+16j, head slice);
  // ASYNCcnt tracks it while we run the QK^T WMMAs below.
#pragma unroll
  for (int it = 0; it < 32; ++it) {
    const int idx = it * 32 + lane;
    const int row = idx >> 3, ch = idx & 7;
    async_g2l_b128(v + (size_t)(r + 16 * row) * DMODEL + h * DHEAD + ch * 8,
                   &v_lds[row][ch * 8]);
  }

  // S = Q K^T over the gathered kv set.
  v8f s[8] = {};
  const _Float16* qbase = q + (size_t)(16 * r) * DMODEL + h * DHEAD;
  const v16h qa0 = ldfrag(qbase,      DMODEL, lane);
  const v16h qa1 = ldfrag(qbase + 32, DMODEL, lane);
#pragma unroll
  for (int t = 0; t < 8; ++t) {
    // B column c of tile t is K row r + 256t + 16c (row stride 16*DMODEL).
    const _Float16* kbase = k + (size_t)(r + 256 * t) * DMODEL + h * DHEAD;
    s[t] = wmma_f16(qa0, ldfrag(kbase,      16 * DMODEL, lane), s[t]);
    s[t] = wmma_f16(qa1, ldfrag(kbase + 32, 16 * DMODEL, lane), s[t]);
  }

  // Softmax (all 128 gathered columns are unmasked). Row m = j + (lane>=16?8:0);
  // column reduction = xor-shuffles 1/2/4/8 (stay within 16-lane halves).
  const float scale = 0.125f;  // 1/sqrt(64)
  float rsum[8];
#pragma unroll
  for (int j = 0; j < 8; ++j) {
    float mm = s[0][j];
#pragma unroll
    for (int t = 1; t < 8; ++t) mm = fmaxf(mm, s[t][j]);
    mm *= scale;
#pragma unroll
    for (int d = 1; d < 16; d <<= 1) mm = fmaxf(mm, __shfl_xor(mm, d, 32));
    float sum = 0.f;
#pragma unroll
    for (int t = 0; t < 8; ++t) {
      const float e = __expf(s[t][j] * scale - mm);
      s[t][j] = e;
      sum += e;
    }
#pragma unroll
    for (int d = 1; d < 16; d <<= 1) sum += __shfl_xor(sum, d, 32);
    rsum[j] = sum;
  }

  // Spill P (f16) to LDS so it can be reloaded in A-fragment layout.
  const int col  = lane & 15;
  const int rofs = (lane & 16) ? 8 : 0;
#pragma unroll
  for (int t = 0; t < 8; ++t)
#pragma unroll
    for (int j = 0; j < 8; ++j)
      p_lds[j + rofs][16 * t + col] = (_Float16)s[t][j];

  wait_async0();     // V tile resident in LDS
  __syncthreads();   // p_lds visible (single wave: orders ds writes vs reads)

  // O = P V
  v8f o[4] = {};
#pragma unroll
  for (int kt = 0; kt < 4; ++kt) {
    const v16h pa = ldfrag(&p_lds[0][0] + 32 * kt, 128, lane);
#pragma unroll
    for (int n2 = 0; n2 < 4; ++n2) {
      union { v16h v; _Float16 e[16]; } u;
      const int kbb = 32 * kt + ((lane & 16) ? 8 : 0);
      const int d   = 16 * n2 + col;
#pragma unroll
      for (int e = 0; e < 8; ++e) {
        u.e[e]     = v_lds[kbb + e][d];
        u.e[e + 8] = v_lds[kbb + 16 + e][d];
      }
      o[n2] = wmma_f16(pa, u.v, o[n2]);
    }
  }

  // Epilogue: divide by row sum, store f16 to attnout (rows 0..255 only).
#pragma unroll
  for (int n2 = 0; n2 < 4; ++n2)
#pragma unroll
    for (int j = 0; j < 8; ++j) {
      const float denom = fmaxf(rsum[j], 1e-30f);
      aout[(size_t)(16 * r + j + rofs) * DMODEL + h * DHEAD + 16 * n2 + col] =
          (_Float16)(o[n2][j] / denom);
    }
}

// ---------------------------------------------------------------------------
// K4: out[0:256] = x[0:256] + attnout @ w_o (f16 WMMA, f32 residual add).
// ---------------------------------------------------------------------------
__global__ __launch_bounds__(128) void gemm_out_kernel(const _Float16* __restrict__ a,
                                                       const _Float16* __restrict__ wto,
                                                       const float* __restrict__ x,
                                                       float* __restrict__ out) {
  const int lane = threadIdx.x & 31;
  const int wave = threadIdx.x >> 5;
  const int m0 = blockIdx.y * 128 + (wave >> 1) * 64;
  const int n0 = blockIdx.x * 128 + (wave & 1) * 64;

  v8f acc[4][4] = {};
  for (int kk = 0; kk < DMODEL; kk += 32) {
    if (kk + 32 < DMODEL) {
      __builtin_prefetch(a + (size_t)m0 * DMODEL + kk + 32, 0, 1);
      __builtin_prefetch(wto + (size_t)n0 * DMODEL + kk + 32, 0, 1);
    }
    v16h af[4];
#pragma unroll
    for (int i = 0; i < 4; ++i)
      af[i] = ldfrag(a + (size_t)(m0 + 16 * i) * DMODEL + kk, DMODEL, lane);
#pragma unroll
    for (int j = 0; j < 4; ++j) {
      v16h b = ldfrag(wto + (size_t)(n0 + 16 * j) * DMODEL + kk, DMODEL, lane);
#pragma unroll
      for (int i = 0; i < 4; ++i)
        acc[i][j] = wmma_f16(af[i], b, acc[i][j]);
    }
  }
  const int col  = lane & 15;
  const int rofs = (lane & 16) ? 8 : 0;
#pragma unroll
  for (int i = 0; i < 4; ++i)
#pragma unroll
    for (int j = 0; j < 4; ++j)
#pragma unroll
      for (int jj = 0; jj < 8; ++jj) {
        const size_t idx =
            (size_t)(m0 + 16 * i + jj + rofs) * DMODEL + (n0 + 16 * j + col);
        out[idx] = x[idx] + acc[i][j][jj];
      }
}

// ---------------------------------------------------------------------------
// K5: fully-masked query rows (256..2047) -> out = x (attention contributes 0).
// ---------------------------------------------------------------------------
__global__ __launch_bounds__(256) void copy_rows_kernel(const float* __restrict__ x,
                                                        float* __restrict__ out) {
  const size_t i = (size_t)blockIdx.x * blockDim.x + threadIdx.x;
  const size_t n = (size_t)(NTOK - NACT) * DMODEL / 4;
  if (i < n) {
    const float4* s = (const float4*)(x + (size_t)NACT * DMODEL);
    float4* d       = (float4*)(out + (size_t)NACT * DMODEL);
    d[i] = s[i];
  }
}

// ---------------------------------------------------------------------------
extern "C" void kernel_launch(void* const* d_in, const int* in_sizes, int n_in,
                              void* d_out, int out_size, void* d_ws, size_t ws_size,
                              hipStream_t stream) {
  const float* x  = (const float*)d_in[0];
  const float* wq = (const float*)d_in[1];
  const float* wk = (const float*)d_in[2];
  const float* wv = (const float*)d_in[3];
  const float* wo = (const float*)d_in[4];
  float* out = (float*)d_out;

  char* ws = (char*)d_ws;
  _Float16* xn   = (_Float16*)(ws);                          //  4 MB: LN(x) f16
  _Float16* wt   = (_Float16*)(ws + (4ull  << 20));          //  8 MB: Wt q,k,v,o f16 (transposed)
  _Float16* qkv  = (_Float16*)(ws + (12ull << 20));          // 12 MB: q,k,v f16 [3][N][D]
  _Float16* aout = (_Float16*)(ws + (24ull << 20));          // .5 MB: attn out rows 0..255

  const size_t WSZ = (size_t)DMODEL * DMODEL;
  const size_t TSZ = (size_t)NTOK * DMODEL;

  ln_rows_kernel<<<NTOK, 256, 0, stream>>>(x, xn);

  dim3 tgrid(DMODEL / 32, DMODEL / 32), tblk(32, 8);
  wtrans_kernel<<<tgrid, tblk, 0, stream>>>(wq, wt + 0 * WSZ);
  wtrans_kernel<<<tgrid, tblk, 0, stream>>>(wk, wt + 1 * WSZ);
  wtrans_kernel<<<tgrid, tblk, 0, stream>>>(wv, wt + 2 * WSZ);
  wtrans_kernel<<<tgrid, tblk, 0, stream>>>(wo, wt + 3 * WSZ);

  gemm_qkv_kernel<<<dim3(DMODEL / 128, NTOK / 128, 3), 128, 0, stream>>>(xn, wt, qkv);

  head_ln_kernel<<<dim3((NTOK * NHEAD) / 8, 2), 256, 0, stream>>>(qkv, qkv + TSZ);

  attention_kernel<<<NHEAD * 16, 32, 0, stream>>>(qkv, qkv + TSZ, qkv + 2 * TSZ, aout);

  gemm_out_kernel<<<dim3(DMODEL / 128, NACT / 128), 128, 0, stream>>>(
      aout, wt + 3 * WSZ, x, out);

  copy_rows_kernel<<<((NTOK - NACT) * DMODEL / 4 + 255) / 256, 256, 0, stream>>>(x, out);
}
